// BranchedAttnProcessor_18476949307901
// MI455X (gfx1250) — matrix operand order repaired and checked by reading, not compiled, not run on hardware
//
#include <hip/hip_runtime.h>
#include <hip/hip_bf16.h>

// MI455X / gfx1250, wave32. All matmuls via v_wmma_f32_16x16x32_bf16.
// Round 3: LDS-staged, double-buffered tiles everywhere.
//  - GEMM: 128x128 block tile, 8 waves, A/B staged via global_load_async_to_lds_b128,
//    double-buffered with s_wait_asynccnt + barrier; wave = 32x64 tile (8 wmma/step).
//  - Flash: block = 128 q-rows (4 waves x 32 rows); K & V tiles (32 keys) staged in
//    shared LDS once per block step (waves 0/1 issue async), double-buffered.
//  - LDS row padding (40/72 bf16) to spread the 16-lane tile reads across banks.
//  - softmax reductions stay on DPP16 ROW_XMASK (pure VALU).

typedef __attribute__((ext_vector_type(16))) __bf16 v16bf;
typedef __attribute__((ext_vector_type(8)))  __bf16 v8bf;
typedef __attribute__((ext_vector_type(4)))  __bf16 v4bf;
typedef __attribute__((ext_vector_type(8)))  float  v8f;

#define CDIM  1280
#define HEADS 20
#define DHEAD 64
#define SQ    2048   // query seq per merged batch (2S)
#define SK    1024   // kv seq per batch

__device__ __forceinline__ v16bf make_a16(v8bf lo, v8bf hi) {
  v16bf a;
#pragma unroll
  for (int e = 0; e < 8; ++e) { a[e] = lo[e]; a[e + 8] = hi[e]; }
  return a;
}
__device__ __forceinline__ v8f wmma_bf16(v16bf a, v16bf b, v8f c) {
  return __builtin_amdgcn_wmma_f32_16x16x32_bf16(false, a, false, b, (short)0, c,
                                                 false, false);
}

// DPP16 ROW_XMASK butterfly within each 16-lane half (pure VALU, no DS waits).
template <int CTRL>
__device__ __forceinline__ float dpp_xmask(float x) {
  return __int_as_float(__builtin_amdgcn_update_dpp(
      0, __float_as_int(x), CTRL, 0xf, 0xf, false));
}
__device__ __forceinline__ float rowmax16(float x) {
  x = fmaxf(x, dpp_xmask<0x161>(x));
  x = fmaxf(x, dpp_xmask<0x162>(x));
  x = fmaxf(x, dpp_xmask<0x164>(x));
  x = fmaxf(x, dpp_xmask<0x168>(x));
  return x;
}
__device__ __forceinline__ float rowsum16(float x) {
  x += dpp_xmask<0x161>(x);
  x += dpp_xmask<0x162>(x);
  x += dpp_xmask<0x164>(x);
  x += dpp_xmask<0x168>(x);
  return x;
}

// ---------------------------------------------------------------------------
// elementwise f32 -> bf16 (one pass; no cvt in GEMM hot loops)
// ---------------------------------------------------------------------------
__global__ __launch_bounds__(256) void cvt_f32_bf16(
    const float* __restrict__ X, __bf16* __restrict__ Y, int n)
{
  const int i = (blockIdx.x * 256 + threadIdx.x) * 4;
  if (i >= n) return;
  float4 v = *(const float4*)(X + i);
  v4bf o;
  o[0] = (__bf16)v.x; o[1] = (__bf16)v.y; o[2] = (__bf16)v.z; o[3] = (__bf16)v.w;
  *(v4bf*)(Y + i) = o;
}

// ---------------------------------------------------------------------------
// GEMM: Y[m,n] = sum_k X[m,k] * W[n,k]  (bf16 in; + optional row gate / bias)
// 256 thr (8 waves); block tile 128x128; wave tile 32x64 (2x4 accums).
// A/B K-slabs (128x32) staged in LDS via async loads, double-buffered.
// A-layout (16-bit 16x32): m = lane%16, k = (e/8)*16 + (lane/16)*8 + (e%8)
// B-layout (16-bit 32x16): n = lane%16, k = (e/8)*16 + (lane/16)*8 + (e%8)
// C/D-layout (f32 16x16):  m = (lane/16)*8 + r, n = lane%16
// ---------------------------------------------------------------------------
template <bool OUT_BF16>
__global__ __launch_bounds__(256) void gemm_bf16(
    const __bf16* __restrict__ X, const __bf16* __restrict__ W,
    const float* __restrict__ gate, const float* __restrict__ bias,
    void* __restrict__ Yv, int M, int N, int K)
{
  const int tid  = threadIdx.x;
  const int lane = tid & 31;
  const int wave = tid >> 5;
  const int half = lane >> 4;
  const int l16  = lane & 15;
  const int m0b = blockIdx.x * 128;
  const int n0b = blockIdx.y * 128;
  const int m_w = (wave & 3) * 32;
  const int n_w = (wave >> 2) * 64;

  constexpr int LDP = 40;   // padded row stride (bf16): 80B -> conflict-free
  __shared__ __attribute__((aligned(32))) __bf16 Ab[2][128 * LDP];
  __shared__ __attribute__((aligned(32))) __bf16 Bb[2][128 * LDP];

  auto stage = [&](int buf, int kk) {
    const bool isB = tid >= 128;
    const int  row = tid & 127;
    const __bf16* src =
        (isB ? W + (size_t)(n0b + row) * K : X + (size_t)(m0b + row) * K) + kk;
    __bf16* dst = (isB ? Bb[buf] : Ab[buf]) + row * LDP;
    const unsigned lo = (unsigned)(uintptr_t)dst;
    const unsigned long long ga = (unsigned long long)(uintptr_t)src;
    asm volatile("global_load_async_to_lds_b128 %0, %1, off"           :: "v"(lo), "v"(ga) : "memory");
    asm volatile("global_load_async_to_lds_b128 %0, %1, off offset:16" :: "v"(lo), "v"(ga) : "memory");
    asm volatile("global_load_async_to_lds_b128 %0, %1, off offset:32" :: "v"(lo), "v"(ga) : "memory");
    asm volatile("global_load_async_to_lds_b128 %0, %1, off offset:48" :: "v"(lo), "v"(ga) : "memory");
  };

  v8f acc[2][4] = {};

  stage(0, 0);
  asm volatile("s_wait_asynccnt 0x0" ::: "memory");
  __syncthreads();

  const int nsteps = K / 32;
  for (int step = 0; step < nsteps; ++step) {
    const int cur = step & 1;
    if (step + 1 < nsteps) stage(1 - cur, (step + 1) * 32);

    v16bf a[2], bt[4];
#pragma unroll
    for (int i = 0; i < 2; ++i) {
      const __bf16* p = Ab[cur] + (m_w + i * 16 + l16) * LDP + half * 8;
      a[i] = make_a16(*(const v8bf*)p, *(const v8bf*)(p + 16));
    }
#pragma unroll
    for (int j = 0; j < 4; ++j) {
      const __bf16* p = Bb[cur] + (n_w + j * 16 + l16) * LDP + half * 8;
      bt[j] = make_a16(*(const v8bf*)p, *(const v8bf*)(p + 16));
    }
#pragma unroll
    for (int i = 0; i < 2; ++i)
#pragma unroll
      for (int j = 0; j < 4; ++j)
        acc[i][j] = wmma_bf16(a[i], bt[j], acc[i][j]);

    if (step + 1 < nsteps) {
      asm volatile("s_wait_asynccnt 0x0" ::: "memory");
      __syncthreads();
    }
  }

#pragma unroll
  for (int i = 0; i < 2; ++i) {
#pragma unroll
    for (int r = 0; r < 8; ++r) {
      const int row = m0b + m_w + i * 16 + half * 8 + r;
      const float g = gate ? gate[row] : 1.0f;
#pragma unroll
      for (int j = 0; j < 4; ++j) {
        const int col = n0b + n_w + j * 16 + l16;
        float v = acc[i][j][r] * g;
        if (OUT_BF16) {
          ((__bf16*)Yv)[(size_t)row * N + col] = (__bf16)v;
        } else {
          ((float*)Yv)[(size_t)row * N + col] = v + (bias ? bias[col] : 0.0f);
        }
      }
    }
  }
}

// ---------------------------------------------------------------------------
// Flash attention, one branch. Q (2x2048xC), K/V (2x1024xC), all bf16
// token-major; head slice = cols [h*64, h*64+64).
// Block = 128 q rows (4 waves x 32 rows = two 16-row WMMA tiles each).
// Per 32-key step: waves 0/1 async-stage K/V tiles into shared LDS,
// double-buffered; all waves consume them. Out (merged) is bf16.
// ---------------------------------------------------------------------------
__global__ __launch_bounds__(128) void flash_attn(
    const __bf16* __restrict__ Qb, const __bf16* __restrict__ Kb,
    const __bf16* __restrict__ Vb, __bf16* __restrict__ Out, int accumulate)
{
  const int lane = threadIdx.x & 31;
  const int wave = threadIdx.x >> 5;
  const int half = lane >> 4;
  const int l16  = lane & 15;
  const int hb = blockIdx.y;
  const int b  = hb / HEADS;
  const int h  = hb % HEADS;
  const int q0 = blockIdx.x * 128 + wave * 32;   // this wave: rows q0..q0+31

  constexpr int LDK = 72;  // padded row stride (bf16) for K/V tiles
  __shared__ __attribute__((aligned(32))) __bf16 kbuf[2][32 * LDK];
  __shared__ __attribute__((aligned(32))) __bf16 vbuf[2][32 * LDK];
  __shared__ __attribute__((aligned(32))) __bf16 pstA[4][2 * 16 * 40];
  __bf16* pst = pstA[wave];

  auto stage = [&](int buf, int k0) {
    if (wave < 2) {
      const __bf16* src = (wave == 0 ? Kb : Vb)
          + (size_t)(b * SK + k0 + lane) * CDIM + h * DHEAD;
      __bf16* dst = (wave == 0 ? kbuf[buf] : vbuf[buf]) + lane * LDK;
      const unsigned lo = (unsigned)(uintptr_t)dst;
      const unsigned long long ga = (unsigned long long)(uintptr_t)src;
      asm volatile("global_load_async_to_lds_b128 %0, %1, off"            :: "v"(lo), "v"(ga) : "memory");
      asm volatile("global_load_async_to_lds_b128 %0, %1, off offset:16"  :: "v"(lo), "v"(ga) : "memory");
      asm volatile("global_load_async_to_lds_b128 %0, %1, off offset:32"  :: "v"(lo), "v"(ga) : "memory");
      asm volatile("global_load_async_to_lds_b128 %0, %1, off offset:48"  :: "v"(lo), "v"(ga) : "memory");
      asm volatile("global_load_async_to_lds_b128 %0, %1, off offset:64"  :: "v"(lo), "v"(ga) : "memory");
      asm volatile("global_load_async_to_lds_b128 %0, %1, off offset:80"  :: "v"(lo), "v"(ga) : "memory");
      asm volatile("global_load_async_to_lds_b128 %0, %1, off offset:96"  :: "v"(lo), "v"(ga) : "memory");
      asm volatile("global_load_async_to_lds_b128 %0, %1, off offset:112" :: "v"(lo), "v"(ga) : "memory");
      asm volatile("s_wait_asynccnt 0x0" ::: "memory");
    }
  };

  // Q A-tiles: 2 row sub-tiles x 2 d-chunks
  v16bf aq[2][2];
#pragma unroll
  for (int u = 0; u < 2; ++u) {
    const __bf16* qp = Qb + (size_t)(b * SQ + q0 + u * 16 + l16) * CDIM + h * DHEAD;
#pragma unroll
    for (int dk = 0; dk < 2; ++dk) {
      const __bf16* qpp = qp + dk * 32 + half * 8;
      aq[u][dk] = make_a16(*(const v8bf*)(qpp), *(const v8bf*)(qpp + 16));
    }
  }

  float mrow[2][8], lrow[2][8];
  v8f o[2][4] = {};
#pragma unroll
  for (int u = 0; u < 2; ++u)
#pragma unroll
    for (int r = 0; r < 8; ++r) { mrow[u][r] = -1e30f; lrow[u][r] = 0.0f; }

  stage(0, 0);
  __syncthreads();

  const int nsteps = SK / 32;
  for (int step = 0; step < nsteps; ++step) {
    const int cur = step & 1;
    if (step + 1 < nsteps) stage(1 - cur, (step + 1) * 32);

    // K B-tiles (shared by both q sub-tiles): j = key sub-tile, dk = d chunk
    v16bf bk[2][2];
#pragma unroll
    for (int j = 0; j < 2; ++j)
#pragma unroll
      for (int dk = 0; dk < 2; ++dk) {
        const __bf16* kp = kbuf[cur] + (j * 16 + l16) * LDK + dk * 32 + half * 8;
        bk[j][dk] = make_a16(*(const v8bf*)kp, *(const v8bf*)(kp + 16));
      }

    // V B-tiles (shared): t = d chunk of 16; gathered column-wise from LDS
    v16bf bv[4];
#pragma unroll
    for (int t = 0; t < 4; ++t) {
      const __bf16* vb0 = vbuf[cur] + (half * 16) * LDK + t * 16 + l16;
#pragma unroll
      for (int e = 0; e < 16; ++e) bv[t][e] = vb0[e * LDK];
    }

#pragma unroll
    for (int u = 0; u < 2; ++u) {
      // S = Q K^T
      v8f s[2] = {};
#pragma unroll
      for (int j = 0; j < 2; ++j)
#pragma unroll
        for (int dk = 0; dk < 2; ++dk)
          s[j] = wmma_bf16(aq[u][dk], bk[j][dk], s[j]);

      // online softmax (row r spans 16 lanes; DPP reductions)
#pragma unroll
      for (int r = 0; r < 8; ++r) {
        float s0 = s[0][r] * 0.125f;   // 1/sqrt(64)
        float s1 = s[1][r] * 0.125f;
        const float mnew  = fmaxf(mrow[u][r], rowmax16(fmaxf(s0, s1)));
        const float alpha = __expf(mrow[u][r] - mnew);
        mrow[u][r] = mnew;
        const float p0 = __expf(s0 - mnew);
        const float p1 = __expf(s1 - mnew);
        s[0][r] = p0; s[1][r] = p1;
        lrow[u][r] = lrow[u][r] * alpha + rowsum16(p0 + p1);
        o[u][0][r] *= alpha; o[u][1][r] *= alpha;
        o[u][2][r] *= alpha; o[u][3][r] *= alpha;
      }

      // P (C/D layout) -> LDS 16x(32 padded to 40), read back in A layout
      __bf16* pu = pst + u * 16 * 40;
#pragma unroll
      for (int j = 0; j < 2; ++j)
#pragma unroll
        for (int r = 0; r < 8; ++r)
          pu[(half * 8 + r) * 40 + j * 16 + l16] = (__bf16)s[j][r];

      const __bf16* pr = pu + l16 * 40 + half * 16;
      v16bf pa = make_a16(*(const v8bf*)pr, *(const v8bf*)(pr + 8));

      // O += P V
#pragma unroll
      for (int t = 0; t < 4; ++t)
        o[u][t] = wmma_bf16(pa, bv[t], o[u][t]);
    }

    if (step + 1 < nsteps) __syncthreads();
  }

  // epilogue: normalize, (optionally) accumulate, store bf16 merged
#pragma unroll
  for (int u = 0; u < 2; ++u)
#pragma unroll
    for (int r = 0; r < 8; ++r) {
      const float inv = 1.0f / lrow[u][r];
      const size_t rowoff =
          (size_t)(b * SQ + q0 + u * 16 + half * 8 + r) * CDIM + h * DHEAD;
#pragma unroll
      for (int t = 0; t < 4; ++t) {
        float v = o[u][t][r] * inv;
        __bf16* op = Out + rowoff + t * 16 + l16;
        if (accumulate) v += (float)*op;
        *op = (__bf16)v;
      }
    }
}

// ---------------------------------------------------------------------------
extern "C" void kernel_launch(void* const* d_in, const int* in_sizes, int n_in,
                              void* d_out, int out_size, void* d_ws, size_t ws_size,
                              hipStream_t stream)
{
  (void)in_sizes; (void)n_in; (void)out_size; (void)ws_size;
  const float* hs   = (const float*)d_in[0];   // (4,1024,1280) f32
  const float* mask = (const float*)d_in[1];   // (2,1024,1)    f32
  const float* Wq   = (const float*)d_in[2];
  const float* Wk   = (const float*)d_in[3];
  const float* Wv   = (const float*)d_in[4];
  const float* Wo   = (const float*)d_in[5];
  const float* bo   = (const float*)d_in[6];
  float* out = (float*)d_out;                  // (2,2048,1280) f32

  char* ws = (char*)d_ws;
  __bf16* Qb  = (__bf16*)ws; ws += (size_t)4096 * CDIM * 2;
  __bf16* Kbg = (__bf16*)ws; ws += (size_t)2048 * CDIM * 2;
  __bf16* Vbg = (__bf16*)ws; ws += (size_t)2048 * CDIM * 2;
  __bf16* Kfa = (__bf16*)ws; ws += (size_t)2048 * CDIM * 2;
  __bf16* Vfa = (__bf16*)ws; ws += (size_t)2048 * CDIM * 2;
  __bf16* merged = (__bf16*)ws; ws += (size_t)4096 * CDIM * 2;
  __bf16* hsb = (__bf16*)ws; ws += (size_t)4096 * CDIM * 2;
  __bf16* Wqb = (__bf16*)ws; ws += (size_t)CDIM * CDIM * 2;
  __bf16* Wkb = (__bf16*)ws; ws += (size_t)CDIM * CDIM * 2;
  __bf16* Wvb = (__bf16*)ws; ws += (size_t)CDIM * CDIM * 2;
  __bf16* Wob = (__bf16*)ws;

  const int nHS = 4096 * CDIM;       // 5,242,880
  const int nW  = CDIM * CDIM;       // 1,638,400

  // one-shot bf16 conversions
  cvt_f32_bf16<<<dim3(nHS / 1024), dim3(256), 0, stream>>>(hs, hsb, nHS);
  cvt_f32_bf16<<<dim3(nW / 1024),  dim3(256), 0, stream>>>(Wq, Wqb, nW);
  cvt_f32_bf16<<<dim3(nW / 1024),  dim3(256), 0, stream>>>(Wk, Wkb, nW);
  cvt_f32_bf16<<<dim3(nW / 1024),  dim3(256), 0, stream>>>(Wv, Wvb, nW);
  cvt_f32_bf16<<<dim3(nW / 1024),  dim3(256), 0, stream>>>(Wo, Wob, nW);

  // projections (bf16 outputs); block tile 128x128, 256 threads
  gemm_bf16<true><<<dim3(32, 10), dim3(256), 0, stream>>>(hsb, Wqb, nullptr, nullptr, Qb, 4096, CDIM, CDIM);
  gemm_bf16<true><<<dim3(16, 10), dim3(256), 0, stream>>>(hsb, Wkb, nullptr, nullptr, Kbg, 2048, CDIM, CDIM);
  gemm_bf16<true><<<dim3(16, 10), dim3(256), 0, stream>>>(hsb, Wvb, nullptr, nullptr, Vbg, 2048, CDIM, CDIM);
  gemm_bf16<true><<<dim3(16, 10), dim3(256), 0, stream>>>(hsb + (size_t)2048 * CDIM, Wkb, mask, nullptr, Kfa, 2048, CDIM, CDIM);
  gemm_bf16<true><<<dim3(16, 10), dim3(256), 0, stream>>>(hsb + (size_t)2048 * CDIM, Wvb, mask, nullptr, Vfa, 2048, CDIM, CDIM);

  // attention branches: bg writes, face accumulates
  flash_attn<<<dim3(16, 40), dim3(128), 0, stream>>>(Qb, Kbg, Vbg, merged, 0);
  flash_attn<<<dim3(16, 40), dim3(128), 0, stream>>>(Qb, Kfa, Vfa, merged, 1);

  // output projection (f32 + bias)
  gemm_bf16<false><<<dim3(32, 10), dim3(256), 0, stream>>>(merged, Wob, nullptr, bo, out, 4096, CDIM, CDIM);
}
